// DeltaAI_84061099918079
// MI455X (gfx1250) — matrix operand order repaired and checked by reading, not compiled, not run on hardware
//
#include <hip/hip_runtime.h>
#include <stdint.h>

#define B_SZ   4096
#define VDIM_  1024
#define XDIM_  128
#define HDIM_  512
#define MAXPA_ 8
#define MAXCH_ 8
#define RROWS  (B_SZ * 9)     /* 36864 qnet rows */
#define LN_EPS 1e-5f

typedef __attribute__((ext_vector_type(16))) __bf16 v16bf;
typedef __attribute__((ext_vector_type(8)))  float  v8f;

struct U32x2 { uint4 a; uint4 b; };
static_assert(sizeof(U32x2) == 32, "frag size");
static_assert(sizeof(v16bf) == 32, "frag size");

static __device__ __forceinline__ float bf2f(uint16_t u) {
  uint32_t x = (uint32_t)u << 16;
  return __builtin_bit_cast(float, x);
}
static __device__ __forceinline__ uint16_t f2bf(float f) {
  uint32_t x = __builtin_bit_cast(uint32_t, f);
  uint32_t r = x + 0x7FFFu + ((x >> 16) & 1u);   // round-to-nearest-even
  return (uint16_t)(r >> 16);
}
static __device__ __forceinline__ float eluf(float x) { return x > 0.f ? x : expm1f(x); }
static __device__ __forceinline__ float logsigf(float x) {
  return fminf(x, 0.f) - log1pf(expf(-fabsf(x)));
}
// 16 contiguous bf16 (32B) -> v16bf
static __device__ __forceinline__ v16bf ld_b_frag(const char* p) {
  uint4 lo = *(const uint4*)p;
  uint4 hi = *(const uint4*)(p + 16);
  return __builtin_bit_cast(v16bf, U32x2{lo, hi});
}

// ---------------------------------------------------------------------------
// Kernel 0: convert W2, W3 (f32 [K,N]) to bf16 transposed [N,K] so that a
// WMMA B-fragment (16 contiguous K values per lane) is a contiguous 32B load.
// ---------------------------------------------------------------------------
__global__ __launch_bounds__(256) void k0_transpose(
    const float* __restrict__ W2, const float* __restrict__ W3,
    uint16_t* __restrict__ W2T, uint16_t* __restrict__ W3T) {
  int idx = blockIdx.x * 256 + threadIdx.x;     // 0 .. 2*512*512-1
  int mat = idx >> 18;
  int rem = idx & ((1 << 18) - 1);
  int n = rem >> 9;
  int k = rem & 511;
  const float* src = mat ? W3 : W2;
  uint16_t*    dst = mat ? W3T : W2T;
  dst[(size_t)n * HDIM_ + k] = f2bf(src[(size_t)k * HDIM_ + n]);
}

// ---------------------------------------------------------------------------
// Kernel 1: layer 1.  V_eff is binary & sparse: union of [0,XDIM) and the
// (deduped, masked) parent set of the row's node.  x1 = b1 + sum of active W1
// rows; then LayerNorm + ELU; store bf16 H1 plus node index and |V_eff| count.
// One workgroup per batch element b; slots s=0..8 share the [0,XDIM) base sum.
// ---------------------------------------------------------------------------
__global__ __launch_bounds__(256) void k1_layer1(
    const float* __restrict__ V, const int* __restrict__ Kpa,
    const int* __restrict__ Kch, const int* __restrict__ ilist,
    const float* __restrict__ W1, const float* __restrict__ b1,
    const float* __restrict__ g1, const float* __restrict__ be1,
    uint16_t* __restrict__ H1, int* __restrict__ nodeArr, int* __restrict__ cntArr) {
  __shared__ float vflag[XDIM_];
  __shared__ int   pinc[MAXPA_];
  __shared__ float redS[8], redQ[8];
  __shared__ float s_mean, s_rstd;
  __shared__ int   s_cntx;

  const int b = blockIdx.x;
  const int t = threadIdx.x;
  const float* vrow = V + (size_t)b * VDIM_;
  if (t < XDIM_) vflag[t] = vrow[t];
  __syncthreads();
  if (t == 0) {
    int c = 0;
    for (int k = 0; k < XDIM_; ++k) c += (vflag[k] != 0.f);
    s_cntx = c;
  }
  __syncthreads();

  const int h0 = t, h1 = t + 256;
  float base0 = 0.f, base1 = 0.f;
  for (int k = 0; k < XDIM_; ++k) {
    if (vflag[k] != 0.f) {
      base0 += W1[(size_t)k * HDIM_ + h0];
      base1 += W1[(size_t)k * HDIM_ + h1];
    }
  }
  const int i0 = ilist[b];

  for (int s = 0; s < 9; ++s) {
    int node;
    if (s == 0) node = i0;
    else { int j = Kch[(size_t)i0 * MAXCH_ + (s - 1)]; node = (j >= 0) ? j : 0; }

    if (t < MAXPA_) {
      int p = Kpa[(size_t)node * MAXPA_ + t];
      int inc = (p >= XDIM_) && (vrow[p] != 0.f);   // p<XDIM already in base set
      for (int a = 0; a < t && inc; ++a)            // scatter-set semantics: dedupe
        if (Kpa[(size_t)node * MAXPA_ + a] == p) inc = 0;
      pinc[t] = inc ? p : -1;
    }
    __syncthreads();

    float a0 = base0, a1 = base1;
    #pragma unroll
    for (int a = 0; a < MAXPA_; ++a) {
      int q = pinc[a];
      if (q >= 0) { a0 += W1[(size_t)q * HDIM_ + h0]; a1 += W1[(size_t)q * HDIM_ + h1]; }
    }
    float x0 = a0 + b1[h0], x1 = a1 + b1[h1];

    // block LayerNorm over 512 values (2 per thread)
    float ps = x0 + x1, pq = x0 * x0 + x1 * x1;
    #pragma unroll
    for (int m = 16; m >= 1; m >>= 1) {
      ps += __shfl_xor(ps, m, 32);
      pq += __shfl_xor(pq, m, 32);
    }
    int wv = t >> 5;
    if ((t & 31) == 0) { redS[wv] = ps; redQ[wv] = pq; }
    __syncthreads();
    if (t == 0) {
      float S = 0.f, Q = 0.f;
      for (int w = 0; w < 8; ++w) { S += redS[w]; Q += redQ[w]; }
      float mean = S / (float)HDIM_;
      s_mean = mean;
      s_rstd = rsqrtf(Q / (float)HDIM_ - mean * mean + LN_EPS);
    }
    __syncthreads();
    float mean = s_mean, rstd = s_rstd;
    float y0 = eluf((x0 - mean) * rstd * g1[h0] + be1[h0]);
    float y1 = eluf((x1 - mean) * rstd * g1[h1] + be1[h1]);
    size_t row = (size_t)b * 9 + s;
    H1[row * HDIM_ + h0] = f2bf(y0);
    H1[row * HDIM_ + h1] = f2bf(y1);
    if (t == 0) {
      int extra = 0;
      for (int a = 0; a < MAXPA_; ++a) extra += (pinc[a] >= 0);
      nodeArr[row] = node;
      cntArr[row] = s_cntx + extra;     // |V_eff| (binary states)
    }
    __syncthreads();   // pinc reused next slot
  }
}

// ---------------------------------------------------------------------------
// Kernel 2: fused layers 2+3 + head, WMMA bf16.
// 512 threads = 16 waves; 64 rows x 512 cols per workgroup.
// wave w: row-tile rt = w&3 (16 rows), column quarter (w>>2)*128 -> 8 tiles.
// 16 waves (4/SIMD) hide LDS/WMMA latency; 8 v8f accumulators per wave keep
// VGPR pressure low.  H tile in LDS (stride 520 bf16, conflict-free A frags).
// W^T chunks (32 K x 512 N, 64B/row) double-buffered in LDS (stride 40 bf16):
// one barrier per K-chunk; global loads of chunk i+1 overlap WMMAs of chunk i.
// B fragments held in a 2-deep rotating register buffer loaded 2 tiles ahead.
// ---------------------------------------------------------------------------
#define MTILE      64
#define A_STRIDE   1040                    /* bytes: 520 bf16 */
#define B_STRIDE   80                      /* bytes: 40 bf16 (64B used) */
#define TILE_STRIDE (16 * B_STRIDE)        /* 1280: 16 N-rows per tile */
#define ABUF_BYTES (MTILE * A_STRIDE)      /* 66560 */
#define BBUF_OFF   ABUF_BYTES
#define BBUF_BYTES (HDIM_ * B_STRIDE)      /* 40960 per buffer */
#define RED_OFF    (BBUF_OFF + 2 * BBUF_BYTES)
#define SMEM_BYTES (RED_OFF + 128 * 4 * 4) /* 150528 total */

__global__ __launch_bounds__(512) void k2_mlp(
    const uint16_t* __restrict__ H1,
    const uint16_t* __restrict__ W2T, const uint16_t* __restrict__ W3T,
    const float* __restrict__ b2, const float* __restrict__ g2, const float* __restrict__ be2,
    const float* __restrict__ b3, const float* __restrict__ g3, const float* __restrict__ be3,
    const float* __restrict__ headW, const float* __restrict__ headb,
    const float* __restrict__ marginals, const float* __restrict__ V,
    const int* __restrict__ nodeArr, const int* __restrict__ cntArr,
    float* __restrict__ ll) {
  extern __shared__ char smem[];
  char*  abuf = smem;
  float* red  = (float*)(smem + RED_OFF);   // [2][64][4] : {sum,sq} x row x quarter

  const int t = threadIdx.x;
  const int row0 = blockIdx.x * MTILE;

  // load H tile -> LDS (bf16 row-major, padded stride); 8 threads per row
  {
    int r = t >> 3;
    int kp = (t & 7) * 64;
    const uint4* src = (const uint4*)(H1 + (size_t)(row0 + r) * HDIM_ + kp);
    #pragma unroll
    for (int i = 0; i < 8; ++i)
      *(uint4*)(abuf + r * A_STRIDE + kp * 2 + i * 16) = src[i];
  }
  __syncthreads();

  const int lane = t & 31;
  const int wv = t >> 5;                 // 0..15
  const int rt = wv & 3;                 // row tile
  const int quarter = wv >> 2;           // 0..3
  const int colbase = quarter * 128;     // 8 col tiles of 16
  const int lm = lane & 15;
  const int lh = lane >> 4;

  #pragma unroll 1
  for (int layer = 0; layer < 2; ++layer) {
    const uint16_t* WT = layer ? W3T : W2T;
    const float* bb = layer ? b3 : b2;
    const float* gg = layer ? g3 : g2;
    const float* ee = layer ? be3 : be2;

    v8f acc[8];
    #pragma unroll
    for (int ct = 0; ct < 8; ++ct) acc[ct] = (v8f){0.f,0.f,0.f,0.f,0.f,0.f,0.f,0.f};

    // prologue: stage chunk 0 (64B per W^T row, one row per thread)
    {
      const uint16_t* wr = WT + (size_t)t * HDIM_;
      uint4 s0 = *(const uint4*)(wr +  0);
      uint4 s1 = *(const uint4*)(wr +  8);
      uint4 s2 = *(const uint4*)(wr + 16);
      uint4 s3 = *(const uint4*)(wr + 24);
      char* bw = smem + BBUF_OFF + t * B_STRIDE;
      *(uint4*)(bw +  0) = s0;
      *(uint4*)(bw + 16) = s1;
      *(uint4*)(bw + 32) = s2;
      *(uint4*)(bw + 48) = s3;
    }
    __syncthreads();

    #pragma unroll 1
    for (int it = 0; it < 16; ++it) {
      const int kc = it * 32;
      // issue global loads for next chunk (in flight during the WMMAs below)
      uint4 s0, s1, s2, s3;
      if (it < 15) {
        const uint16_t* wr = WT + (size_t)t * HDIM_ + kc + 32;
        s0 = *(const uint4*)(wr +  0);
        s1 = *(const uint4*)(wr +  8);
        s2 = *(const uint4*)(wr + 16);
        s3 = *(const uint4*)(wr + 24);
      }

      // A fragment: lane lm -> row rt*16+lm; K = kc + lh*16 + {0..15}
      const char* ap = abuf + (rt * 16 + lm) * A_STRIDE + (kc + lh * 16) * 2;
      uint4 alo = *(const uint4*)ap;
      uint4 ahi = *(const uint4*)(ap + 16);
      v16bf afrag = __builtin_bit_cast(v16bf, U32x2{alo, ahi});

      // B fragments: 2-deep rotating register buffer, loaded 2 tiles ahead
      const char* bbase = smem + BBUF_OFF + (size_t)(it & 1) * BBUF_BYTES
                          + (colbase + lm) * B_STRIDE + lh * 32;
      v16bf bfr[2];
      bfr[0] = ld_b_frag(bbase);
      bfr[1] = ld_b_frag(bbase + TILE_STRIDE);
      #pragma unroll
      for (int ct = 0; ct < 8; ++ct) {
        acc[ct] = __builtin_amdgcn_wmma_f32_16x16x32_bf16(
            false, afrag, false, bfr[ct & 1], (short)0, acc[ct], false, false);
        if (ct + 2 < 8) bfr[ct & 1] = ld_b_frag(bbase + (ct + 2) * TILE_STRIDE);
      }

      // stage next chunk into the other LDS buffer (its readers finished at
      // the barrier ending iteration it-1)
      if (it < 15) {
        char* bw = smem + BBUF_OFF + (size_t)((it + 1) & 1) * BBUF_BYTES
                   + t * B_STRIDE;
        *(uint4*)(bw +  0) = s0;
        *(uint4*)(bw + 16) = s1;
        *(uint4*)(bw + 32) = s2;
        *(uint4*)(bw + 48) = s3;
      }
      __syncthreads();
    }

    // bias before LN statistics
    #pragma unroll
    for (int ct = 0; ct < 8; ++ct) {
      float bv = bb[colbase + ct * 16 + lm];
      #pragma unroll
      for (int r = 0; r < 8; ++r) acc[ct][r] += bv;
    }

    // per-row partial sums over this wave's 128 columns
    float ps[8], pq[8];
    #pragma unroll
    for (int r = 0; r < 8; ++r) { ps[r] = 0.f; pq[r] = 0.f; }
    #pragma unroll
    for (int ct = 0; ct < 8; ++ct)
      #pragma unroll
      for (int r = 0; r < 8; ++r) { float x = acc[ct][r]; ps[r] += x; pq[r] += x * x; }
    #pragma unroll
    for (int r = 0; r < 8; ++r) {
      #pragma unroll
      for (int m = 1; m <= 8; m <<= 1) {
        ps[r] += __shfl_xor(ps[r], m, 32);
        pq[r] += __shfl_xor(pq[r], m, 32);
      }
    }
    if (lm == 0) {
      #pragma unroll
      for (int r = 0; r < 8; ++r) {
        int row = rt * 16 + lh * 8 + r;
        red[row * 4 + quarter]        = ps[r];
        red[(64 + row) * 4 + quarter] = pq[r];
      }
    }
    __syncthreads();
    float mean[8], rstd[8];
    #pragma unroll
    for (int r = 0; r < 8; ++r) {
      int row = rt * 16 + lh * 8 + r;
      float S = red[row * 4 + 0] + red[row * 4 + 1] + red[row * 4 + 2] + red[row * 4 + 3];
      float Q = red[(64 + row) * 4 + 0] + red[(64 + row) * 4 + 1]
              + red[(64 + row) * 4 + 2] + red[(64 + row) * 4 + 3];
      float m = S / (float)HDIM_;
      mean[r] = m;
      rstd[r] = rsqrtf(Q / (float)HDIM_ - m * m + LN_EPS);
    }

    // LN + ELU + residual; write back bf16 into abuf (disjoint elements/lane)
    #pragma unroll 1
    for (int ct = 0; ct < 8; ++ct) {
      int n = colbase + ct * 16 + lm;
      float gv = gg[n], ev = ee[n];
      #pragma unroll
      for (int r = 0; r < 8; ++r) {
        int row = rt * 16 + lh * 8 + r;
        uint16_t* pe = (uint16_t*)(abuf + row * A_STRIDE + n * 2);
        float hprev = bf2f(*pe);
        float y = eluf((acc[ct][r] - mean[r]) * rstd[r] * gv + ev);
        *pe = f2bf(hprev + y);
      }
    }
    __syncthreads();
  }

  // head: 8 threads per row, 64 elements each, then shfl reduce
  {
    int r = t >> 3;
    int part = t & 7;
    size_t grow = (size_t)row0 + r;
    int node = nodeArr[grow];
    const float*    hw   = headW + (size_t)node * HDIM_ + part * 64;
    const uint16_t* hrow = (const uint16_t*)(abuf + r * A_STRIDE) + part * 64;
    float s = 0.f;
    for (int k = 0; k < 64; ++k) s += bf2f(hrow[k]) * hw[k];
    s += __shfl_xor(s, 1, 32);
    s += __shfl_xor(s, 2, 32);
    s += __shfl_xor(s, 4, 32);
    if (part == 0) {
      float logit = (cntArr[grow] == 0) ? marginals[node] : (s + headb[node]);
      int b = (int)(grow / 9);
      float target = V[(size_t)b * VDIM_ + node];
      ll[grow] = target * logsigf(logit) + (1.f - target) * logsigf(-logit);
    }
  }
}

// ---------------------------------------------------------------------------
// Kernel 3: deterministic fixed-order combine -> out[2,B]
// ---------------------------------------------------------------------------
__global__ __launch_bounds__(256) void k3_combine(
    const float* __restrict__ ll, const int* __restrict__ Kch,
    const int* __restrict__ ilist, float* __restrict__ out) {
  int b = blockIdx.x * 256 + threadIdx.x;
  if (b >= B_SZ) return;
  int i = ilist[b];
  out[b] = ll[(size_t)b * 9];
  float s = 0.f;
  for (int c = 0; c < MAXCH_; ++c)
    if (Kch[(size_t)i * MAXCH_ + c] >= 0) s += ll[(size_t)b * 9 + 1 + c];
  out[B_SZ + b] = s;
}

// ---------------------------------------------------------------------------
extern "C" void kernel_launch(void* const* d_in, const int* in_sizes, int n_in,
                              void* d_out, int out_size, void* d_ws, size_t ws_size,
                              hipStream_t stream) {
  const float* V     = (const float*)d_in[0];
  const int*   Kpa   = (const int*)d_in[1];
  const int*   Kch   = (const int*)d_in[2];
  const int*   il    = (const int*)d_in[3];
  const float* W1    = (const float*)d_in[4];
  const float* b1    = (const float*)d_in[5];
  const float* g1    = (const float*)d_in[6];
  const float* be1   = (const float*)d_in[7];
  const float* W2    = (const float*)d_in[8];
  const float* b2    = (const float*)d_in[9];
  const float* g2    = (const float*)d_in[10];
  const float* be2   = (const float*)d_in[11];
  const float* W3    = (const float*)d_in[12];
  const float* b3    = (const float*)d_in[13];
  const float* g3    = (const float*)d_in[14];
  const float* be3   = (const float*)d_in[15];
  const float* headW = (const float*)d_in[16];
  const float* headb = (const float*)d_in[17];
  const float* marg  = (const float*)d_in[18];

  char* ws = (char*)d_ws;
  uint16_t* H1      = (uint16_t*)(ws + 0);                 // 36864*512*2 = 37748736
  uint16_t* W2T     = (uint16_t*)(ws + 37748736);          // 524288
  uint16_t* W3T     = (uint16_t*)(ws + 38273024);          // 524288
  int*      nodeArr = (int*)     (ws + 38797312);          // 147456
  int*      cntArr  = (int*)     (ws + 38944768);          // 147456
  float*    ll      = (float*)   (ws + 39092224);          // 147456
  float*    out     = (float*)d_out;

  k0_transpose<<<2048, 256, 0, stream>>>(W2, W3, W2T, W3T);
  k1_layer1<<<B_SZ, 256, 0, stream>>>(V, Kpa, Kch, il, W1, b1, g1, be1, H1, nodeArr, cntArr);
  k2_mlp<<<RROWS / MTILE, 512, SMEM_BYTES, stream>>>(
      H1, W2T, W3T, b2, g2, be2, b3, g3, be3, headW, headb, marg, V, nodeArr, cntArr, ll);
  k3_combine<<<(B_SZ + 255) / 256, 256, 0, stream>>>(ll, Kch, il, out);
}
